// sLSTMblock_5557687681257
// MI455X (gfx1250) — matrix operand order repaired and checked by reading, not compiled, not run on hardware
//
#include <hip/hip_runtime.h>
#include <hip/hip_bf16.h>
#include <stdint.h>

typedef __bf16 bf16;
typedef bf16  v16bf __attribute__((ext_vector_type(16)));
typedef float v8f   __attribute__((ext_vector_type(8)));

#if defined(__AMDGCN__) && __has_builtin(__builtin_amdgcn_tensor_load_to_lds) && __has_builtin(__builtin_amdgcn_s_wait_tensorcnt)
#define USE_TDM 1
#else
#define USE_TDM 0
#endif

typedef unsigned int u32x4 __attribute__((ext_vector_type(4)));
typedef int          i32x8 __attribute__((ext_vector_type(8)));
typedef int          i32x4 __attribute__((ext_vector_type(4)));

constexpr int B_   = 4;
constexpr int T_   = 2048;
constexpr int D_   = 1024;
constexpr int H_   = 8;
constexpr int DH_  = 128;
constexpr int K_   = 4;
constexpr int BT_  = B_ * T_;      // 8192
constexpr int DK_  = D_ * K_;      // 4096 (im2col K)
constexpr int DFF_ = 1365;
constexpr int DFFP_= 1408;         // padded to 11 * 128
constexpr int TS_  = 36;           // LDS tile row stride in bf16 (padded from 32)

// ---------------------------------------------------------------------------
// WMMA fragment loaders (ISA 7.12.2 layouts, 16-bit operands, wave32)
// ---------------------------------------------------------------------------
__device__ __forceinline__ v16bf frag_a(const bf16* t, int lane) {
  v16bf f;
  const int r = lane & 15, hf = lane >> 4;
  const bf16* row = t + r * TS_;
#pragma unroll
  for (int v = 0; v < 8; ++v) {
    int k = ((v >> 2) << 4) + (hf << 3) + ((v & 3) << 1);
    f[2 * v]     = row[k];
    f[2 * v + 1] = row[k + 1];
  }
  return f;
}

__device__ __forceinline__ v16bf frag_b(const bf16* t, int lane) {
  v16bf f;
  const int n = lane & 15, hf = lane >> 4;
  const bf16* row = t + n * TS_;
#pragma unroll
  for (int v = 0; v < 8; ++v) {
    int k = ((v >> 2) << 4) + (hf << 3) + ((v & 3) << 1);
    f[2 * v]     = row[k];
    f[2 * v + 1] = row[k + 1];
  }
  return f;
}

// One wave computes a 16(M) x 128(N) strip: 8 wmma with a reused A fragment.
__device__ __forceinline__ void wmma_tile(const bf16* As, const bf16* Bs,
                                          v8f* acc, int wv, int lane) {
  v16bf a = frag_a(As + wv * 16 * TS_, lane);
#pragma unroll
  for (int nt = 0; nt < 8; ++nt) {
    v16bf b = frag_b(Bs + nt * 16 * TS_, lane);
    acc[nt] = __builtin_amdgcn_wmma_f32_16x16x32_bf16(
        false, a, false, b, (short)0, acc[nt], false, false);
  }
}

// Copy 16 contiguous bf16 (32B) as 4x u64; both sides 8B aligned by layout.
__device__ __forceinline__ void cp16(bf16* dst, const bf16* src) {
  const unsigned long long* s = (const unsigned long long*)src;
  unsigned long long* d = (unsigned long long*)dst;
  d[0] = s[0]; d[1] = s[1]; d[2] = s[2]; d[3] = s[3];
}
__device__ __forceinline__ void zero16(bf16* dst) {
  unsigned long long* d = (unsigned long long*)dst;
  d[0] = 0ull; d[1] = 0ull; d[2] = 0ull; d[3] = 0ull;
}

__device__ __forceinline__ float block_sum(float v, float* red) {
  const int lt = threadIdx.x;
  __syncthreads();
  red[lt] = v;
  __syncthreads();
  for (int o = 128; o > 0; o >>= 1) {
    if (lt < o) red[lt] += red[lt + o];
    __syncthreads();
  }
  return red[0];
}

// ---------------------------------------------------------------------------
// Weight repacks (fp32 -> bf16, layout changes so GEMM K is contiguous)
// ---------------------------------------------------------------------------
__global__ void repack_conv_kernel(const float* __restrict__ cw, bf16* __restrict__ wc) {
  int idx = blockIdx.x * 256 + threadIdx.x;          // over D*DK
  if (idx >= D_ * DK_) return;
  int o = idx >> 12, r = idx & 4095;
  int tap = r >> 10, i = r & 1023;
  wc[idx] = (bf16)cw[(size_t)(o * D_ + i) * K_ + tap];   // [o][tap][i]
}

__global__ void repack_gates_kernel(const float* __restrict__ Wz, const float* __restrict__ Wi,
                                    const float* __restrict__ Wf, const float* __restrict__ Wo,
                                    bf16* __restrict__ wt) {
  int idx = blockIdx.x * 256 + threadIdx.x;          // over 4*H*128*128
  if (idx >= 4 * H_ * DH_ * DH_) return;
  int d = idx & 127, e = (idx >> 7) & 127, hh = (idx >> 14) & 7, g = idx >> 17;
  const float* W = (g == 0) ? Wz : (g == 1) ? Wi : (g == 2) ? Wf : Wo;
  wt[idx] = (bf16)W[(size_t)(hh * DH_ + d) * DH_ + e];   // [g][h][e][d]
}

__global__ void padcast_kernel(const float* __restrict__ src, bf16* __restrict__ dst,
                               int rd, int cd, int rs, int cs) {
  int idx = blockIdx.x * 256 + threadIdx.x;
  if (idx >= rd * cd) return;
  int r = idx / cd, c = idx - r * cd;
  dst[idx] = (r < rs && c < cs) ? (bf16)src[(size_t)r * cs + c] : (bf16)0.0f;
}

// ---------------------------------------------------------------------------
// LayerNorm (x -> bf16 x_norm)
// ---------------------------------------------------------------------------
__global__ void ln_kernel(const float* __restrict__ x, const float* __restrict__ g,
                          const float* __restrict__ b, bf16* __restrict__ out) {
  __shared__ float red[256];
  const int row = blockIdx.x, lt = threadIdx.x;
  const float* xr = x + (size_t)row * D_;
  float v[4], s = 0.f;
#pragma unroll
  for (int i = 0; i < 4; ++i) { v[i] = xr[lt + 256 * i]; s += v[i]; }
  float mu = block_sum(s, red) * (1.f / D_);
  s = 0.f;
#pragma unroll
  for (int i = 0; i < 4; ++i) { float d = v[i] - mu; s += d * d; }
  float rs = rsqrtf(block_sum(s, red) * (1.f / D_) + 1e-5f);
  bf16* orow = out + (size_t)row * D_;
#pragma unroll
  for (int i = 0; i < 4; ++i) {
    int c = lt + 256 * i;
    orow[c] = (bf16)((v[i] - mu) * rs * g[c] + b[c]);
  }
}

// GN(h) + x -> h_skip (f32), LN(h_skip) -> hn (bf16)
__global__ void post_ln_kernel(const float* __restrict__ hseq, const float* __restrict__ x,
                               const float* __restrict__ gng, const float* __restrict__ gnb,
                               const float* __restrict__ lng, const float* __restrict__ lnb,
                               float* __restrict__ hskip, bf16* __restrict__ hn) {
  __shared__ float red[256];
  const int row = blockIdx.x, lt = threadIdx.x;
  const float* hr = hseq + (size_t)row * D_;
  const float* xr = x + (size_t)row * D_;
  float v[4], s = 0.f;
#pragma unroll
  for (int i = 0; i < 4; ++i) { v[i] = hr[lt + 256 * i]; s += v[i]; }
  float mu = block_sum(s, red) * (1.f / D_);
  s = 0.f;
#pragma unroll
  for (int i = 0; i < 4; ++i) { float d = v[i] - mu; s += d * d; }
  float rs = rsqrtf(block_sum(s, red) * (1.f / D_) + 1e-5f);
  float w[4]; s = 0.f;
#pragma unroll
  for (int i = 0; i < 4; ++i) {
    int c = lt + 256 * i;
    w[i] = (v[i] - mu) * rs * gng[c] + gnb[c] + xr[c];
    hskip[(size_t)row * D_ + c] = w[i];
    s += w[i];
  }
  float mu2 = block_sum(s, red) * (1.f / D_);
  s = 0.f;
#pragma unroll
  for (int i = 0; i < 4; ++i) { float d = w[i] - mu2; s += d * d; }
  float rs2 = rsqrtf(block_sum(s, red) * (1.f / D_) + 1e-5f);
#pragma unroll
  for (int i = 0; i < 4; ++i) {
    int c = lt + 256 * i;
    hn[(size_t)row * D_ + c] = (bf16)((w[i] - mu2) * rs2 * lng[c] + lnb[c]);
  }
}

// ---------------------------------------------------------------------------
// Conv-as-GEMM (+bias +swish) -> x_if (bf16).  M=BT, N=D, K=4*D (4 taps)
// ---------------------------------------------------------------------------
__global__ void conv_gemm_kernel(const bf16* __restrict__ xn, const bf16* __restrict__ wc,
                                 const float* __restrict__ cb, bf16* __restrict__ xif) {
  __shared__ bf16 As[128 * TS_];
  __shared__ bf16 Bs[128 * TS_];
  const int m0 = blockIdx.x * 128, n0 = blockIdx.y * 128;
  const int bb = m0 >> 11, t0 = m0 & (T_ - 1);
  const int lt = threadIdx.x;
  const int srow = lt >> 1, scol = (lt & 1) << 4;
  const int lane = lt & 31, wv = lt >> 5;
  v8f acc[8] = {};
  for (int tap = 0; tap < K_; ++tap) {
    for (int kb = 0; kb < D_ / 32; ++kb) {
      __syncthreads();
      int tsrc = t0 + srow - (K_ - 1) + tap;
      bf16* ad = As + srow * TS_ + scol;
      if (tsrc >= 0) cp16(ad, xn + (size_t)(bb * T_ + tsrc) * D_ + kb * 32 + scol);
      else zero16(ad);
      cp16(Bs + srow * TS_ + scol,
           wc + (size_t)(n0 + srow) * DK_ + tap * D_ + kb * 32 + scol);
      __syncthreads();
      wmma_tile(As, Bs, acc, wv, lane);
    }
  }
  const int hf = lane >> 4, lc = lane & 15;
#pragma unroll
  for (int nt = 0; nt < 8; ++nt)
#pragma unroll
    for (int r = 0; r < 8; ++r) {
      int mm = m0 + wv * 16 + r + hf * 8;
      int col = n0 + nt * 16 + lc;
      float v = acc[nt][r] + cb[col];
      float sw = v / (1.f + expf(-v));           // swish
      xif[(size_t)mm * D_ + col] = (bf16)sw;
    }
}

// ---------------------------------------------------------------------------
// Block-diagonal gate projections: w{z,i,f,o} (f32).  grid (BT/128, H, 4)
// ---------------------------------------------------------------------------
__global__ void proj_gemm_kernel(const bf16* __restrict__ xn, const bf16* __restrict__ xif,
                                 const bf16* __restrict__ wt,
                                 const float* __restrict__ bz, const float* __restrict__ bi,
                                 const float* __restrict__ bfg, const float* __restrict__ bo,
                                 float* __restrict__ wbar) {
  __shared__ bf16 As[128 * TS_];
  __shared__ bf16 Bs[128 * TS_];
  const int m0 = blockIdx.x * 128;
  const int hh = blockIdx.y;
  const int g  = blockIdx.z;
  const bf16* A = (g == 0 || g == 3) ? xn : xif;
  const float* bias = (g == 0) ? bz : (g == 1) ? bi : (g == 2) ? bfg : bo;
  const int lt = threadIdx.x;
  const int srow = lt >> 1, scol = (lt & 1) << 4;
  const int lane = lt & 31, wv = lt >> 5;
  v8f acc[8] = {};
  for (int kb = 0; kb < DH_ / 32; ++kb) {
    __syncthreads();
    cp16(As + srow * TS_ + scol,
         A + (size_t)(m0 + srow) * D_ + hh * DH_ + kb * 32 + scol);
    cp16(Bs + srow * TS_ + scol,
         wt + ((size_t)(g * H_ + hh) * DH_ + srow) * DH_ + kb * 32 + scol);
    __syncthreads();
    wmma_tile(As, Bs, acc, wv, lane);
  }
  const int hf = lane >> 4, lc = lane & 15;
  const size_t gstride = (size_t)BT_ * D_;
#pragma unroll
  for (int nt = 0; nt < 8; ++nt)
#pragma unroll
    for (int r = 0; r < 8; ++r) {
      int mm = m0 + wv * 16 + r + hf * 8;
      int dim = hh * DH_ + nt * 16 + lc;
      wbar[(size_t)g * gstride + (size_t)mm * D_ + dim] = acc[nt][r] + bias[dim];
    }
}

// ---------------------------------------------------------------------------
// sLSTM scan: one block per (batch, head); R matrices resident in LDS (TDM).
// ---------------------------------------------------------------------------
__global__ void scan_kernel(const float* __restrict__ wbar,
                            const float* __restrict__ Rz, const float* __restrict__ Ri,
                            const float* __restrict__ Rf, const float* __restrict__ Ro,
                            float* __restrict__ hseq, float* __restrict__ dout) {
  extern __shared__ float smem[];
  float* Rs   = smem;                   // [4][128][128]
  float* gbar = smem + 4 * DH_ * DH_;   // [4][128]
  float* hvec = gbar + 512;             // [128]
  const int bb = blockIdx.x >> 3;
  const int hh = blockIdx.x & 7;
  const int lt = threadIdx.x;
  const int g = lt >> 7, j = lt & 127;
  const float* Rptr[4] = { Rz + (size_t)hh * DH_ * DH_, Ri + (size_t)hh * DH_ * DH_,
                           Rf + (size_t)hh * DH_ * DH_, Ro + (size_t)hh * DH_ * DH_ };
#if USE_TDM
  if (lt < 32) {
    unsigned lbase = (unsigned)(uintptr_t)(void*)Rs;       // LDS offset = low 32 bits
    for (int gg = 0; gg < 4; ++gg) {
      unsigned long long ga = (unsigned long long)(uintptr_t)Rptr[gg];
      // D# group0: count=1 | lds_addr | global_addr | type=2
      u32x4 g0 = { 1u, lbase + (unsigned)(gg * 65536),
                   (unsigned)(ga & 0xFFFFFFFFull),
                   (unsigned)(ga >> 32) | (2u << 30) };
      // D# group1: data_size=4B; 1D tile of 16384 elements
      i32x8 g1 = { (int)(2u << 16),          // workgroup_mask=0, data_size=2(4B)
                   (int)(16384u << 16),      // tensor_dim0[15:0]
                   (int)(1u << 16),          // tensor_dim0[31:16]=0 | tensor_dim1[15:0]=1
                   (int)(16384u << 16),      // tensor_dim1 hi=0 | tile_dim0=16384
                   0,                        // tile_dim1=0 (unused), tile_dim2=0
                   16384, 0, 0 };            // tensor_dim0_stride
      i32x4 gz4 = { 0, 0, 0, 0 };
      i32x8 gz8 = { 0, 0, 0, 0, 0, 0, 0, 0 };
      __builtin_amdgcn_tensor_load_to_lds(g0, g1, gz4, gz4, gz8, 0);
    }
    __builtin_amdgcn_s_wait_tensorcnt(0);
  }
#else
  for (int idx = lt; idx < DH_ * DH_; idx += 512) {
    Rs[0 * 16384 + idx] = Rptr[0][idx];
    Rs[1 * 16384 + idx] = Rptr[1][idx];
    Rs[2 * 16384 + idx] = Rptr[2][idx];
    Rs[3 * 16384 + idx] = Rptr[3][idx];
  }
#endif
  if (lt < 128) hvec[lt] = 0.f;
  __syncthreads();

  float c = 0.f, n = 0.f, m = 0.f, hreg = 0.f;
  const size_t gstride = (size_t)BT_ * D_;
  const float* wb = wbar + (size_t)g * gstride + (size_t)(bb * T_) * D_ + hh * DH_ + j;
  const float* Rg = Rs + g * 16384 + j;   // Rs[g][k][j], k stride 128
  float* hout = hseq + (size_t)(bb * T_) * D_ + hh * DH_ + j;

  for (int t = 0; t < T_; ++t) {
    float acc = wb[(size_t)t * D_];
#pragma unroll 8
    for (int k = 0; k < DH_; ++k)
      acc = fmaf(hvec[k], Rg[k * 128], acc);
    gbar[lt] = acc;
    __syncthreads();
    if (lt < 128) {
      float zb = gbar[j], ib = gbar[128 + j], fb = gbar[256 + j], ob = gbar[384 + j];
      float mn = fmaxf(fb + m, ib);
      float iv = expf(ib - mn);
      float fv = expf(fb + m - mn);
      c = fv * c + iv * tanhf(zb);
      n = fv * n + iv;
      m = mn;
      float o = 1.f / (1.f + expf(-ob));
      hreg = o * (c / fmaxf(fabsf(n), 1e-8f));
      hvec[j] = hreg;
      hout[(size_t)t * D_] = hreg;
    }
    __syncthreads();
  }
  if (lt < 128) {
    float* outs = dout + (size_t)BT_ * D_;
    size_t p = (size_t)bb * D_ + hh * DH_ + j;
    outs[p] = hreg;
    outs[(size_t)B_ * D_ + p] = c;
    outs[2 * (size_t)B_ * D_ + p] = n;
    outs[3 * (size_t)B_ * D_ + p] = m;
  }
}

// ---------------------------------------------------------------------------
// GLU: zg = (hn @ Lw^T + lb) * gelu(hn @ Rw^T + rb)  -> bf16 [BT, DFFP]
// ---------------------------------------------------------------------------
__global__ void glu_gemm_kernel(const bf16* __restrict__ hn, const bf16* __restrict__ lw,
                                const bf16* __restrict__ rw, const float* __restrict__ lb,
                                const float* __restrict__ rb, bf16* __restrict__ zg) {
  __shared__ bf16 As[128 * TS_];
  __shared__ bf16 Bl[128 * TS_];
  __shared__ bf16 Br[128 * TS_];
  const int m0 = blockIdx.x * 128, n0 = blockIdx.y * 128;
  const int lt = threadIdx.x;
  const int srow = lt >> 1, scol = (lt & 1) << 4;
  const int lane = lt & 31, wv = lt >> 5;
  v8f accl[8] = {}, accr[8] = {};
  for (int kb = 0; kb < D_ / 32; ++kb) {
    __syncthreads();
    cp16(As + srow * TS_ + scol, hn + (size_t)(m0 + srow) * D_ + kb * 32 + scol);
    cp16(Bl + srow * TS_ + scol, lw + (size_t)(n0 + srow) * D_ + kb * 32 + scol);
    cp16(Br + srow * TS_ + scol, rw + (size_t)(n0 + srow) * D_ + kb * 32 + scol);
    __syncthreads();
    wmma_tile(As, Bl, accl, wv, lane);
    wmma_tile(As, Br, accr, wv, lane);
  }
  const int hf = lane >> 4, lc = lane & 15;
#pragma unroll
  for (int nt = 0; nt < 8; ++nt)
#pragma unroll
    for (int r = 0; r < 8; ++r) {
      int mm = m0 + wv * 16 + r + hf * 8;
      int col = n0 + nt * 16 + lc;
      float lbv = (col < DFF_) ? lb[col] : 0.f;
      float rbv = (col < DFF_) ? rb[col] : 0.f;
      float l = accl[nt][r] + lbv;
      float rr = accr[nt][r] + rbv;
      float gel = 0.5f * rr * (1.f + erff(rr * 0.70710678118654752f));
      zg[(size_t)mm * DFFP_ + col] = (bf16)(l * gel);
    }
}

// ---------------------------------------------------------------------------
// Final GEMM: out = zg @ last_w^T + last_b + h_skip   (f32 output)
// ---------------------------------------------------------------------------
__global__ void last_gemm_kernel(const bf16* __restrict__ zg, const bf16* __restrict__ lwt,
                                 const float* __restrict__ lastb,
                                 const float* __restrict__ hskip, float* __restrict__ out) {
  __shared__ bf16 As[128 * TS_];
  __shared__ bf16 Bs[128 * TS_];
  const int m0 = blockIdx.x * 128, n0 = blockIdx.y * 128;
  const int lt = threadIdx.x;
  const int srow = lt >> 1, scol = (lt & 1) << 4;
  const int lane = lt & 31, wv = lt >> 5;
  v8f acc[8] = {};
  for (int kb = 0; kb < DFFP_ / 32; ++kb) {
    __syncthreads();
    cp16(As + srow * TS_ + scol, zg + (size_t)(m0 + srow) * DFFP_ + kb * 32 + scol);
    cp16(Bs + srow * TS_ + scol, lwt + (size_t)(n0 + srow) * DFFP_ + kb * 32 + scol);
    __syncthreads();
    wmma_tile(As, Bs, acc, wv, lane);
  }
  const int hf = lane >> 4, lc = lane & 15;
#pragma unroll
  for (int nt = 0; nt < 8; ++nt)
#pragma unroll
    for (int r = 0; r < 8; ++r) {
      int mm = m0 + wv * 16 + r + hf * 8;
      int col = n0 + nt * 16 + lc;
      out[(size_t)mm * D_ + col] = acc[nt][r] + lastb[col] + hskip[(size_t)mm * D_ + col];
    }
}

// ---------------------------------------------------------------------------
extern "C" void kernel_launch(void* const* d_in, const int* in_sizes, int n_in,
                              void* d_out, int out_size, void* d_ws, size_t ws_size,
                              hipStream_t stream) {
  (void)in_sizes; (void)n_in; (void)out_size; (void)ws_size;
  const float* x      = (const float*)d_in[0];
  const float* Wz     = (const float*)d_in[1];
  const float* Wi     = (const float*)d_in[2];
  const float* Wf     = (const float*)d_in[3];
  const float* Wo     = (const float*)d_in[4];
  const float* bz     = (const float*)d_in[5];
  const float* bi     = (const float*)d_in[6];
  const float* bfg    = (const float*)d_in[7];
  const float* bo     = (const float*)d_in[8];
  const float* Rz     = (const float*)d_in[9];
  const float* Ri     = (const float*)d_in[10];
  const float* Rf     = (const float*)d_in[11];
  const float* Ro     = (const float*)d_in[12];
  const float* conv_w = (const float*)d_in[13];
  const float* conv_b = (const float*)d_in[14];
  const float* ln_g   = (const float*)d_in[15];
  const float* ln_b   = (const float*)d_in[16];
  const float* gn_g   = (const float*)d_in[17];
  const float* gn_b   = (const float*)d_in[18];
  const float* left_w = (const float*)d_in[19];
  const float* left_b = (const float*)d_in[20];
  const float* right_w= (const float*)d_in[21];
  const float* right_b= (const float*)d_in[22];
  const float* last_w = (const float*)d_in[23];
  const float* last_b = (const float*)d_in[24];
  float* out = (float*)d_out;

  size_t off = 0;
  char* wsb = (char*)d_ws;
  auto nxt = [&](size_t bytes) {
    void* p = wsb + off;
    off += (bytes + 255) & ~(size_t)255;
    return p;
  };
  bf16*  xn    = (bf16*) nxt((size_t)BT_ * D_ * 2);
  bf16*  xif   = (bf16*) nxt((size_t)BT_ * D_ * 2);
  bf16*  wc    = (bf16*) nxt((size_t)D_ * DK_ * 2);
  bf16*  wt    = (bf16*) nxt((size_t)4 * H_ * DH_ * DH_ * 2);
  bf16*  lw    = (bf16*) nxt((size_t)DFFP_ * D_ * 2);
  bf16*  rw    = (bf16*) nxt((size_t)DFFP_ * D_ * 2);
  bf16*  lwt   = (bf16*) nxt((size_t)D_ * DFFP_ * 2);
  float* wbar  = (float*)nxt((size_t)4 * BT_ * D_ * 4);
  float* hseq  = (float*)nxt((size_t)BT_ * D_ * 4);
  float* hskip = (float*)nxt((size_t)BT_ * D_ * 4);
  bf16*  hn    = (bf16*) nxt((size_t)BT_ * D_ * 2);
  bf16*  zg    = (bf16*) nxt((size_t)BT_ * DFFP_ * 2);

  // 1) input LN + weight repacks (independent)
  ln_kernel<<<BT_, 256, 0, stream>>>(x, ln_g, ln_b, xn);
  repack_conv_kernel<<<(D_ * DK_ + 255) / 256, 256, 0, stream>>>(conv_w, wc);
  repack_gates_kernel<<<(4 * H_ * DH_ * DH_ + 255) / 256, 256, 0, stream>>>(Wz, Wi, Wf, Wo, wt);
  padcast_kernel<<<(DFFP_ * D_ + 255) / 256, 256, 0, stream>>>(left_w,  lw,  DFFP_, D_, DFF_, D_);
  padcast_kernel<<<(DFFP_ * D_ + 255) / 256, 256, 0, stream>>>(right_w, rw,  DFFP_, D_, DFF_, D_);
  padcast_kernel<<<(D_ * DFFP_ + 255) / 256, 256, 0, stream>>>(last_w,  lwt, D_, DFFP_, D_, DFF_);

  // 2) causal conv as GEMM + swish
  conv_gemm_kernel<<<dim3(BT_ / 128, D_ / 128), 256, 0, stream>>>(xn, wc, conv_b, xif);

  // 3) gate input projections
  proj_gemm_kernel<<<dim3(BT_ / 128, H_, 4), 256, 0, stream>>>(xn, xif, wt, bz, bi, bfg, bo, wbar);

  // 4) sequential sLSTM scan; R matrices LDS-resident (TDM load), 256KB+ dyn LDS
  const size_t scan_lds = (size_t)4 * DH_ * DH_ * 4 + 512 * 4 + 128 * 4;
  scan_kernel<<<B_ * H_, 512, scan_lds, stream>>>(wbar, Rz, Ri, Rf, Ro, hseq, out);

  // 5) GN + skip + LN
  post_ln_kernel<<<BT_, 256, 0, stream>>>(hseq, x, gn_g, gn_b, ln_g, ln_b, hskip, hn);

  // 6) GLU FFN
  glu_gemm_kernel<<<dim3(BT_ / 128, DFFP_ / 128), 256, 0, stream>>>(hn, lw, rw, left_b, right_b, zg);

  // 7) final projection + residual
  last_gemm_kernel<<<dim3(BT_ / 128, D_ / 128), 256, 0, stream>>>(zg, lwt, last_b, hskip, out);
}
